// Net_37391985279546
// MI455X (gfx1250) — compile-verified, matrix operand order checked by reading, not run on hardware
//
#include <hip/hip_runtime.h>
#include <hip/hip_bf16.h>
#include <math.h>

// ---------------------------------------------------------------------------
// GNN pipeline for MI455X (gfx1250, wave32).
// deg==5 for every node (k=4 in-edges + self loop) so GCN = (self+4 nbrs)/5 + b.
// BatchNorm folded to per-channel scale/shift computed by deterministic
// two-stage reductions (fixed partial slots, no atomics, no memset needed).
// WMMA usage:
//   * conv1 (x[N,40] @ W[40,3]) : V_WMMA_F32_16X16X4_F32 (fp32-exact),
//     B/D zero-padded in LDS so fragment builds are branch/exec-free.
//   * head1 [G,32]@[32,16] and head2 [G,16]@[16,128]: V_WMMA_F32_16X16X32_F16.
// ---------------------------------------------------------------------------

#define NPG 128           // nodes per graph
#define EPSBN 1e-5f
#define XWS 17            // padded xw row stride (bank-conflict friendly)

typedef __attribute__((ext_vector_type(16))) _Float16 v16h;
typedef __attribute__((ext_vector_type(8)))  float    v8f;
typedef __attribute__((ext_vector_type(2)))  float    v2f;

#if __has_builtin(__builtin_amdgcn_wmma_f32_16x16x4_f32)
#define HAS_WMMA_F32X4 1
#else
#define HAS_WMMA_F32X4 0
#endif

// ---------------- K1: per-graph KNN + conv1 + mlp1 block1 (lin+relu) -------
__global__ __launch_bounds__(NPG) void k_knn_conv1(
    const float* __restrict__ x, const float* __restrict__ pos,
    const float* __restrict__ W1, const float* __restrict__ b1,
    const float* __restrict__ Wm, const float* __restrict__ bm,
    int* __restrict__ nbr, float* __restrict__ a1)
{
  const int g = blockIdx.x, i = threadIdx.x;
  const int node = g * NPG + i;

  __shared__ float px[NPG], py[NPG], pz[NPG];
  __shared__ float xs[NPG * 40];
  __shared__ float w1p[40 * 16];         // W1 zero-padded to 16 output cols
  __shared__ float xw[NPG * XWS];        // conv1 result, padded row stride

  float p0 = pos[node * 3 + 0], p1 = pos[node * 3 + 1], p2 = pos[node * 3 + 2];
  px[i] = p0; py[i] = p1; pz[i] = p2;
  for (int s = i; s < NPG * 40; s += NPG) xs[s] = x[g * NPG * 40 + s];
  for (int s = i; s < 40 * 16; s += NPG) {
    const int row = s >> 4, col = s & 15;
    w1p[s] = (col < 3) ? W1[row * 3 + col] : 0.f;   // race-free padded fill
  }
  __syncthreads();

  // ---- KNN: 4 nearest (smallest d2), self excluded; branchless insertion --
  float bd0 = 3.8e38f, bd1 = 3.8e38f, bd2 = 3.8e38f, bd3 = 3.8e38f;
  int   bi0 = 0, bi1 = 0, bi2 = 0, bi3 = 0;
  for (int j = 0; j < NPG; ++j) {
    float dx = p0 - px[j], dy = p1 - py[j], dz = p2 - pz[j];
    float d = dx * dx + dy * dy + dz * dz;
    d = (j == i) ? 3.9e38f : d;          // exclude self
    int jj = j;
    bool c;
    c = d < bd0; { float td = c ? bd0 : d; int tj = c ? bi0 : jj;
                   bd0 = c ? d : bd0;     bi0 = c ? jj : bi0; d = td; jj = tj; }
    c = d < bd1; { float td = c ? bd1 : d; int tj = c ? bi1 : jj;
                   bd1 = c ? d : bd1;     bi1 = c ? jj : bi1; d = td; jj = tj; }
    c = d < bd2; { float td = c ? bd2 : d; int tj = c ? bi2 : jj;
                   bd2 = c ? d : bd2;     bi2 = c ? jj : bi2; d = td; jj = tj; }
    c = d < bd3; { bd3 = c ? d : bd3;     bi3 = c ? jj : bi3; }
  }
  nbr[node * 4 + 0] = bi0; nbr[node * 4 + 1] = bi1;
  nbr[node * 4 + 2] = bi2; nbr[node * 4 + 3] = bi3;

  // ---- conv1 xw = x @ W1 (128x40 @ 40x3, cols padded with zeros) ----
#if HAS_WMMA_F32X4
  {
    const int wv = i >> 5, lane = i & 31;
    const int half = lane >> 4, lr = lane & 15;
    // B fragments are row-tile invariant: preload all 10 (K=40 / 4).
    v2f Bf[10];
    #pragma unroll
    for (int kb4 = 0; kb4 < 10; ++kb4) {
      const int k = kb4 * 4 + 2 * half;
      Bf[kb4].x = w1p[k * 16 + lr];
      Bf[kb4].y = w1p[(k + 1) * 16 + lr];
    }
    #pragma unroll
    for (int t = 0; t < 2; ++t) {
      const int m0 = (wv * 2 + t) * 16;
      v8f c = {};
      #pragma unroll
      for (int kb4 = 0; kb4 < 10; ++kb4) {
        const int k = kb4 * 4 + 2 * half;          // A 16x4 fp32 layout
        v2f a;
        a.x = xs[(m0 + lr) * 40 + k];
        a.y = xs[(m0 + lr) * 40 + k + 1];
        c = __builtin_amdgcn_wmma_f32_16x16x4_f32(
            false, a, false, Bf[kb4], (short)0, c, false, false);
      }
      #pragma unroll
      for (int v = 0; v < 8; ++v) {                // C/D: VGPR v -> M = v + 8*half
        const int m = m0 + v + 8 * half;
        if (lr < XWS) xw[m * XWS + lr] = c[v];     // cols>=3 are exact zeros
      }
    }
  }
#else
  {
    float acc0 = 0.f, acc1 = 0.f, acc2 = 0.f;
    for (int k = 0; k < 40; ++k) {
      float xv = xs[i * 40 + k];
      acc0 += xv * w1p[k * 16 + 0];
      acc1 += xv * w1p[k * 16 + 1];
      acc2 += xv * w1p[k * 16 + 2];
    }
    xw[i * XWS + 0] = acc0; xw[i * XWS + 1] = acc1; xw[i * XWS + 2] = acc2;
  }
#endif
  __syncthreads();

  // ---- GCN aggregate (deg==5) + mlp1 block1 linear+relu ----
  float s0 = xw[i*XWS+0], s1 = xw[i*XWS+1], s2 = xw[i*XWS+2];
  s0 += xw[bi0*XWS+0] + xw[bi1*XWS+0] + xw[bi2*XWS+0] + xw[bi3*XWS+0];
  s1 += xw[bi0*XWS+1] + xw[bi1*XWS+1] + xw[bi2*XWS+1] + xw[bi3*XWS+1];
  s2 += xw[bi0*XWS+2] + xw[bi1*XWS+2] + xw[bi2*XWS+2] + xw[bi3*XWS+2];
  float x1g[3] = { s0 * 0.2f + b1[0], s1 * 0.2f + b1[1], s2 * 0.2f + b1[2] };
  #pragma unroll
  for (int o = 0; o < 3; ++o) {
    float v = bm[o];
    #pragma unroll
    for (int c = 0; c < 3; ++c) v += x1g[c] * Wm[c * 3 + o];
    a1[node * 3 + o] = fmaxf(v, 0.f);
  }
}

// ---------------- stats: per-(channel,slice) partial sums (deterministic) ---
__global__ __launch_bounds__(256) void k_stats_col(
    const float* __restrict__ a, int rows, int C, int S, float* __restrict__ partial)
{
  const int c = blockIdx.x / S, s = blockIdx.x % S;
  const int chunk = (rows + S - 1) / S;
  const int r0 = s * chunk, r1 = min(rows, r0 + chunk);
  float sum = 0.f, sq = 0.f;
  for (int r = r0 + threadIdx.x; r < r1; r += 256) {
    float v = a[(long)r * C + c];
    sum += v; sq += v * v;
  }
  __shared__ float ls[256], lq[256];
  ls[threadIdx.x] = sum; lq[threadIdx.x] = sq; __syncthreads();
  for (int off = 128; off > 0; off >>= 1) {
    if (threadIdx.x < off) { ls[threadIdx.x] += ls[threadIdx.x+off]; lq[threadIdx.x] += lq[threadIdx.x+off]; }
    __syncthreads();
  }
  if (threadIdx.x == 0) { partial[(c*S+s)*2] = ls[0]; partial[(c*S+s)*2+1] = lq[0]; }
}

// stage-6 variant: sum and sumsq come from separate per-graph arrays
__global__ __launch_bounds__(256) void k_stats_pair(
    const float* __restrict__ gs, const float* __restrict__ gq,
    int rows, int C, float* __restrict__ partial)
{
  const int c = blockIdx.x;
  float sum = 0.f, sq = 0.f;
  for (int r = threadIdx.x; r < rows; r += 256) { sum += gs[r*C+c]; sq += gq[r*C+c]; }
  __shared__ float ls[256], lq[256];
  ls[threadIdx.x] = sum; lq[threadIdx.x] = sq; __syncthreads();
  for (int off = 128; off > 0; off >>= 1) {
    if (threadIdx.x < off) { ls[threadIdx.x] += ls[threadIdx.x+off]; lq[threadIdx.x] += lq[threadIdx.x+off]; }
    __syncthreads();
  }
  if (threadIdx.x == 0) { partial[c*2] = ls[0]; partial[c*2+1] = lq[0]; }
}

__global__ __launch_bounds__(128) void k_stats_fin(
    const float* __restrict__ partial, int C, int S, float rows,
    const float* __restrict__ gamma, const float* __restrict__ beta,
    float* __restrict__ ss)
{
  const int c = threadIdx.x;
  if (c >= C) return;
  float sum = 0.f, sq = 0.f;
  for (int s = 0; s < S; ++s) { sum += partial[(c*S+s)*2]; sq += partial[(c*S+s)*2+1]; }
  float m = sum / rows;
  float var = sq / rows - m * m;
  float sc = gamma[c] * rsqrtf(var + EPSBN);
  ss[c] = sc;
  ss[C + c] = beta[c] - m * sc;
}

// ---------------- generic BN(fold) + Linear + ReLU (tiny channels) ---------
template<int CI, int CO>
__global__ __launch_bounds__(256) void k_bn_lin(
    const float* __restrict__ ain, const float* __restrict__ ss,
    const float* __restrict__ W, const float* __restrict__ b,
    float* __restrict__ aout, int N)
{
  const int n = blockIdx.x * 256 + threadIdx.x;
  if (n >= N) return;
  float h[CI];
  #pragma unroll
  for (int c = 0; c < CI; ++c) h[c] = ain[(long)n*CI+c] * ss[c] + ss[CI+c];
  #pragma unroll
  for (int o = 0; o < CO; ++o) {
    float v = b[o];
    #pragma unroll
    for (int c = 0; c < CI; ++c) v += h[c] * W[c*CO+o];
    aout[(long)n*CO+o] = fmaxf(v, 0.f);
  }
}

// ---------------- K4: BN(a3)->relu->x1, conv2 gather, mlp2 block1 ----------
__global__ __launch_bounds__(NPG) void k_conv2(
    const float* __restrict__ a3, const float* __restrict__ ss3,
    const float* __restrict__ W2, const float* __restrict__ b2,
    const float* __restrict__ Wm, const float* __restrict__ bm,
    const int* __restrict__ nbr, float* __restrict__ x1s, float* __restrict__ a4)
{
  const int g = blockIdx.x, i = threadIdx.x;
  const int node = g * NPG + i;
  __shared__ float xw[NPG * 6];          // xw2 padded stride 6
  float x1[5];
  #pragma unroll
  for (int c = 0; c < 5; ++c) {
    float h = a3[node*5+c] * ss3[c] + ss3[5+c];
    x1[c] = fmaxf(h, 0.f);
    x1s[node*5+c] = x1[c];
  }
  #pragma unroll
  for (int o = 0; o < 5; ++o) {
    float v = 0.f;
    #pragma unroll
    for (int c = 0; c < 5; ++c) v += x1[c] * W2[c*5+o];
    xw[i*6+o] = v;
  }
  __syncthreads();
  int j0 = nbr[node*4+0], j1 = nbr[node*4+1], j2 = nbr[node*4+2], j3 = nbr[node*4+3];
  float x2g[5];
  #pragma unroll
  for (int o = 0; o < 5; ++o) {
    float s = xw[i*6+o] + xw[j0*6+o] + xw[j1*6+o] + xw[j2*6+o] + xw[j3*6+o];
    x2g[o] = s * 0.2f + b2[o];
  }
  #pragma unroll
  for (int o = 0; o < 5; ++o) {
    float v = bm[o];
    #pragma unroll
    for (int c = 0; c < 5; ++c) v += x2g[c] * Wm[c*5+o];
    a4[node*5+o] = fmaxf(v, 0.f);
  }
}

// ---------------- K6: BN(a5)->relu->x2, concat(x1,x2), line1 lin+relu, -----
// ---------------- per-graph sum & sumsq (pooling + BN6 stats inputs) -------
__global__ __launch_bounds__(NPG) void k_line1(
    const float* __restrict__ a5, const float* __restrict__ ss5,
    const float* __restrict__ x1s,
    const float* __restrict__ Wl, const float* __restrict__ bl,
    float* __restrict__ gsum, float* __restrict__ gsq)
{
  const int g = blockIdx.x, i = threadIdx.x;
  const int node = g * NPG + i;
  __shared__ float av[NPG * 32];
  float z[8];
  #pragma unroll
  for (int c = 0; c < 5; ++c) z[c] = x1s[node*5+c];
  #pragma unroll
  for (int c = 0; c < 3; ++c) {
    float h = a5[node*3+c] * ss5[c] + ss5[3+c];
    z[5+c] = fmaxf(h, 0.f);
  }
  #pragma unroll
  for (int o = 0; o < 32; ++o) {
    float v = bl[o];
    #pragma unroll
    for (int c = 0; c < 8; ++c) v += z[c] * Wl[c*32+o];
    av[i*32+o] = fmaxf(v, 0.f);
  }
  __syncthreads();
  if (i < 32) {
    float s = 0.f, q = 0.f;
    for (int r = 0; r < NPG; ++r) { float v = av[r*32+i]; s += v; q += v*v; }
    gsum[g*32+i] = s; gsq[g*32+i] = q;
  }
}

// pooled = BN6(mean over graph)  (BN affine commutes with mean)
__global__ __launch_bounds__(256) void k_pool(
    const float* __restrict__ gsum, const float* __restrict__ ss6,
    float* __restrict__ pooled, int total)
{
  const int t = blockIdx.x * 256 + threadIdx.x;
  if (t >= total) return;
  const int c = t & 31;
  pooled[t] = (gsum[t] * (1.f / (float)NPG)) * ss6[c] + ss6[32+c];
}

// ---------------- head1: [G,32]@[32,16] with v_wmma_f32_16x16x32_f16 -------
__global__ __launch_bounds__(128) void k_head1(
    const float* __restrict__ pooled, const float* __restrict__ W,
    const float* __restrict__ b, float* __restrict__ out)
{
  const int wv = threadIdx.x >> 5, lane = threadIdx.x & 31;
  const int half = lane >> 4, lr = lane & 15;
  const int r0 = blockIdx.x * 64 + wv * 16;
  v16h A, B;
  #pragma unroll
  for (int v = 0; v < 8; ++v) {
    const int k = ((v < 4) ? 0 : 16) + 2 * (v & 3) + 8 * half;
    A[2*v]   = (_Float16)pooled[(r0 + lr) * 32 + k];
    A[2*v+1] = (_Float16)pooled[(r0 + lr) * 32 + k + 1];
    B[2*v]   = (_Float16)W[k * 16 + lr];
    B[2*v+1] = (_Float16)W[(k + 1) * 16 + lr];
  }
  v8f c = {};
  c = __builtin_amdgcn_wmma_f32_16x16x32_f16(false, A, false, B, (short)0, c, false, false);
  #pragma unroll
  for (int v = 0; v < 8; ++v) {
    const int m = r0 + v + 8 * half;
    out[m * 16 + lr] = fmaxf(c[v] + b[lr], 0.f);
  }
}

// ---------------- head2: BN(ah1) -> [G,16]@[16,128], K padded to 32 --------
__global__ __launch_bounds__(128) void k_head2(
    const float* __restrict__ ah1, const float* __restrict__ ss,
    const float* __restrict__ W, const float* __restrict__ b,
    float* __restrict__ ah2)
{
  const int wv = threadIdx.x >> 5, lane = threadIdx.x & 31;
  const int half = lane >> 4, lr = lane & 15;
  const int r0 = blockIdx.x * 64 + wv * 16;
  v16h A;
  #pragma unroll
  for (int v = 0; v < 8; ++v) {
    const int k = ((v < 4) ? 0 : 16) + 2 * (v & 3) + 8 * half;
    float a0 = (k < 16)     ? ah1[(r0+lr)*16 + k]     * ss[k]   + ss[16+k]   : 0.f;
    float a1 = (k + 1 < 16) ? ah1[(r0+lr)*16 + k + 1] * ss[k+1] + ss[16+k+1] : 0.f;
    A[2*v] = (_Float16)a0; A[2*v+1] = (_Float16)a1;
  }
  for (int n0 = 0; n0 < 128; n0 += 16) {
    v16h B;
    #pragma unroll
    for (int v = 0; v < 8; ++v) {
      const int k = ((v < 4) ? 0 : 16) + 2 * (v & 3) + 8 * half;
      B[2*v]   = (k < 16)     ? (_Float16)W[k * 128 + n0 + lr]       : (_Float16)0.f;
      B[2*v+1] = (k + 1 < 16) ? (_Float16)W[(k + 1) * 128 + n0 + lr] : (_Float16)0.f;
    }
    v8f c = {};
    c = __builtin_amdgcn_wmma_f32_16x16x32_f16(false, A, false, B, (short)0, c, false, false);
    #pragma unroll
    for (int v = 0; v < 8; ++v) {
      const int m = r0 + v + 8 * half;
      ah2[m * 128 + n0 + lr] = fmaxf(c[v] + b[n0 + lr], 0.f);
    }
  }
}

// ---------------- final: BN(ah2) -> @ Wout[128,1] + b -> sigmoid -----------
__global__ __launch_bounds__(256) void k_out(
    const float* __restrict__ ah2, const float* __restrict__ ss,
    const float* __restrict__ Wo, const float* __restrict__ bo,
    float* __restrict__ out, int G)
{
  const int g = blockIdx.x * 256 + threadIdx.x;
  if (g >= G) return;
  float acc = bo[0];
  for (int k = 0; k < 128; ++k)
    acc += (ah2[g*128+k] * ss[k] + ss[128+k]) * Wo[k];
  out[g] = 1.f / (1.f + expf(-acc));
}

// ---------------------------------------------------------------------------
// Host launcher.
// Input flattening assumption: top-level dict insertion order (x, pos, batch,
// params..., num_graphs); params flattened as a jax pytree (dict keys sorted
// alphabetically; within a block dict: W < b < beta < gamma):
//  3:conv1.W 4:conv1.b 5:conv2.W 6:conv2.b
//  7..10:  head_mlp[0] W,b,beta,gamma      11..14: head_mlp[1] W,b,beta,gamma
//  15:head_out.W 16:head_out.b             17..20: line1[0] W,b,beta,gamma
//  21..24: mlp1[0]   25..28: mlp1[1]   29..32: mlp1[2]
//  33..36: mlp2[0]   37..40: mlp2[1]   41: num_graphs
// ---------------------------------------------------------------------------
extern "C" void kernel_launch(void* const* d_in, const int* in_sizes, int n_in,
                              void* d_out, int out_size, void* d_ws, size_t ws_size,
                              hipStream_t stream) {
  const float* x    = (const float*)d_in[0];
  const float* pos  = (const float*)d_in[1];
  const float* f[42];
  for (int i = 0; i < n_in && i < 42; ++i) f[i] = (const float*)d_in[i];

  const int N = in_sizes[0] / 40;     // 524288
  const int G = N / NPG;              // 4096

  char* w = (char*)d_ws;
  auto alloc = [&](size_t bytes) -> void* {
    void* p = (void*)w;
    w += (bytes + 255) & ~(size_t)255;
    return p;
  };
  int*   nbr    = (int*)  alloc((size_t)N * 4 * sizeof(int));
  float* a1     = (float*)alloc((size_t)N * 3 * sizeof(float));
  float* a2     = (float*)alloc((size_t)N * 5 * sizeof(float));
  float* a3     = (float*)alloc((size_t)N * 5 * sizeof(float));
  float* x1s    = (float*)alloc((size_t)N * 5 * sizeof(float));
  float* a4     = (float*)alloc((size_t)N * 5 * sizeof(float));
  float* a5     = (float*)alloc((size_t)N * 3 * sizeof(float));
  float* gsum   = (float*)alloc((size_t)G * 32 * sizeof(float));
  float* gsq    = (float*)alloc((size_t)G * 32 * sizeof(float));
  float* pooled = (float*)alloc((size_t)G * 32 * sizeof(float));
  float* ah1    = (float*)alloc((size_t)G * 16 * sizeof(float));
  float* ah2    = (float*)alloc((size_t)G * 128 * sizeof(float));
  float* part   = (float*)alloc(4096 * sizeof(float));
  float* ss1 = (float*)alloc(512 * sizeof(float));
  float* ss2 = (float*)alloc(512 * sizeof(float));
  float* ss3 = (float*)alloc(512 * sizeof(float));
  float* ss4 = (float*)alloc(512 * sizeof(float));
  float* ss5 = (float*)alloc(512 * sizeof(float));
  float* ss6 = (float*)alloc(512 * sizeof(float));
  float* ss7 = (float*)alloc(512 * sizeof(float));
  float* ss8 = (float*)alloc(512 * sizeof(float));

  const int S = 64;               // slices for node-level stats
  const float fN = (float)N, fG = (float)G;
  const int gridN = (N + 255) / 256;

  // Stage 1: knn + conv1 + mlp1[0] lin/relu
  k_knn_conv1<<<G, NPG, 0, stream>>>(x, pos, f[3], f[4], f[21], f[22], nbr, a1);
  k_stats_col<<<3 * S, 256, 0, stream>>>(a1, N, 3, S, part);
  k_stats_fin<<<1, 128, 0, stream>>>(part, 3, S, fN, f[24], f[23], ss1);

  // Stage 2: mlp1[1] 3->5
  k_bn_lin<3,5><<<gridN, 256, 0, stream>>>(a1, ss1, f[25], f[26], a2, N);
  k_stats_col<<<5 * S, 256, 0, stream>>>(a2, N, 5, S, part);
  k_stats_fin<<<1, 128, 0, stream>>>(part, 5, S, fN, f[28], f[27], ss2);

  // Stage 3: mlp1[2] 5->5
  k_bn_lin<5,5><<<gridN, 256, 0, stream>>>(a2, ss2, f[29], f[30], a3, N);
  k_stats_col<<<5 * S, 256, 0, stream>>>(a3, N, 5, S, part);
  k_stats_fin<<<1, 128, 0, stream>>>(part, 5, S, fN, f[32], f[31], ss3);

  // Stage 4: x1 = relu(BN(a3)); conv2 gather; mlp2[0] 5->5
  k_conv2<<<G, NPG, 0, stream>>>(a3, ss3, f[5], f[6], f[33], f[34], nbr, x1s, a4);
  k_stats_col<<<5 * S, 256, 0, stream>>>(a4, N, 5, S, part);
  k_stats_fin<<<1, 128, 0, stream>>>(part, 5, S, fN, f[36], f[35], ss4);

  // Stage 5: mlp2[1] 5->3
  k_bn_lin<5,3><<<gridN, 256, 0, stream>>>(a4, ss4, f[37], f[38], a5, N);
  k_stats_col<<<3 * S, 256, 0, stream>>>(a5, N, 3, S, part);
  k_stats_fin<<<1, 128, 0, stream>>>(part, 3, S, fN, f[40], f[39], ss5);

  // Stage 6: x2 = relu(BN(a5)); concat; line1 8->32; per-graph sums
  k_line1<<<G, NPG, 0, stream>>>(a5, ss5, x1s, f[17], f[18], gsum, gsq);
  k_stats_pair<<<32, 256, 0, stream>>>(gsum, gsq, G, 32, part);
  k_stats_fin<<<1, 128, 0, stream>>>(part, 32, 1, fN, f[20], f[19], ss6);

  // Pool (BN affine commutes with per-graph mean)
  k_pool<<<(G * 32 + 255) / 256, 256, 0, stream>>>(gsum, ss6, pooled, G * 32);

  // Head1: [G,32]@[32,16] WMMA f16
  k_head1<<<G / 64, 128, 0, stream>>>(pooled, f[7], f[8], ah1);
  k_stats_col<<<16 * 8, 256, 0, stream>>>(ah1, G, 16, 8, part);
  k_stats_fin<<<1, 128, 0, stream>>>(part, 16, 8, fG, f[10], f[9], ss7);

  // Head2: [G,16]@[16,128] WMMA f16 (K padded)
  k_head2<<<G / 64, 128, 0, stream>>>(ah1, ss7, f[11], f[12], ah2);
  k_stats_col<<<128 * 8, 256, 0, stream>>>(ah2, G, 128, 8, part);
  k_stats_fin<<<1, 128, 0, stream>>>(part, 128, 8, fG, f[14], f[13], ss8);

  // Output: sigmoid(BN(ah2) @ Wout + b)
  k_out<<<(G + 255) / 256, 256, 0, stream>>>(ah2, ss8, f[15], f[16], (float*)d_out, G);
}
